// MegaNeRF_5669356832921
// MI455X (gfx1250) — matrix operand order repaired and checked by reading, not compiled, not run on hardware
//
#include <hip/hip_runtime.h>
#include <hip/hip_bf16.h>
#include <math.h>

// Problem constants (from reference): E=8, IN=90, H=256, OUT=4, N=131072
#define NUM_E    8
#define IN_F     90
#define K1P      96      // IN padded to multiple of 32
#define HD       256
#define OUT_D    4
#define NPTS     131072
#define XROW     93      // 3 + IN floats per point row
#define TILE     128     // points per block
#define SF       104     // feats LDS row stride (halves), 96 + 8 pad -> 16B bank skew
#define SH       264     // hidden/weight LDS row stride (halves), 256 + 8 pad
#define NTHREADS 512     // 16 wave32 waves
#define MARGIN_F 1.5f

typedef __attribute__((ext_vector_type(16))) __bf16 v16bf;
typedef __attribute__((ext_vector_type(8)))  float  v8f;

__device__ __forceinline__ unsigned short f2bf(float f) {
  unsigned int u = __float_as_uint(f);
  u += 0x7FFFu + ((u >> 16) & 1u);            // round-to-nearest-even
  return (unsigned short)(u >> 16);
}
__device__ __forceinline__ float bf2f(unsigned short h) {
  return __uint_as_float(((unsigned int)h) << 16);
}

__device__ __forceinline__ v16bf ldfrag(const unsigned short* p0, const unsigned short* p1) {
  union { uint4 q[2]; v16bf v; } u;
  u.q[0] = *(const uint4*)p0;   // ds_load_b128
  u.q[1] = *(const uint4*)p1;   // ds_load_b128
  return u.v;
}

// One dense layer: hout[128 x 256] = relu(hin[128 x kdim] * W + bias)
// wt is the transposed weight panel in LDS: wt[n*SH + k] (N-major, K contiguous)
// 16 waves: wave w owns output N-tile w; 8 accumulators cover all 8 M-tiles,
// so the B fragment is reused across the whole M dimension.
__device__ __forceinline__ void mlp_layer(const unsigned short* __restrict__ hin, int sIn, int kdim,
                                          const unsigned short* __restrict__ wt,
                                          const float* __restrict__ bias,
                                          unsigned short* __restrict__ hout,
                                          int tid, bool relu) {
  const int wave = tid >> 5;          // 0..15 -> N-tile
  const int lane = tid & 31;
  const int l15  = lane & 15;
  const int hi   = lane >> 4;         // lane group (ISA 16-bit fragment split)
  const int nCol = (wave << 4) + l15; // output column owned by this lane

  v8f acc[8];
#pragma unroll
  for (int i = 0; i < 8; ++i) acc[i] = (v8f){0.f,0.f,0.f,0.f,0.f,0.f,0.f,0.f};

  const int kblocks = kdim >> 5;
  for (int kb = 0; kb < kblocks; ++kb) {
    // B fragment (32x16): lane = N, 16 contiguous K halves at hi*16
    const unsigned short* bp = wt + nCol * SH + (kb << 5) + (hi << 4);
    v16bf bfrag = ldfrag(bp, bp + 8);
#pragma unroll
    for (int mt = 0; mt < 8; ++mt) {
      // A fragment (16x32): lane = M row; 16 contiguous K halves at hi*16
      const unsigned short* ap = hin + ((mt << 4) + l15) * sIn + (kb << 5) + (hi << 4);
      v16bf afrag = ldfrag(ap, ap + 8);
      acc[mt] = __builtin_amdgcn_wmma_f32_16x16x32_bf16(
          false, afrag, false, bfrag, (short)0, acc[mt], false, false);
    }
  }

  const float bv = bias[nCol];
#pragma unroll
  for (int mt = 0; mt < 8; ++mt) {
    const int mbase = (mt << 4) + (hi << 3);   // C/D layout: VGPR r -> M = mbase + r
#pragma unroll
    for (int r = 0; r < 8; ++r) {
      float v = acc[mt][r] + bv;
      if (relu) v = fmaxf(v, 0.f);
      hout[(mbase + r) * SH + nCol] = f2bf(v);
    }
  }
}

// Load W[kreal x 256] fp32 (global, row-major) transposed+converted into
// LDS panel wt[n*SH + k], zero-filling k in [kreal, kfill).
__device__ __forceinline__ void load_wT(const float* __restrict__ W, int kreal, int kfill,
                                        unsigned short* __restrict__ wt, int tid) {
  const int cpr = HD / 4;   // float4 chunks per row = 64
  for (int i = tid; i < kfill * cpr; i += NTHREADS) {
    const int k  = i / cpr;
    const int o4 = i - k * cpr;
    float4 v = make_float4(0.f, 0.f, 0.f, 0.f);
    if (k < kreal) v = ((const float4*)(W + (size_t)k * HD))[o4];  // coalesced b128
    const int n = o4 << 2;
    wt[(n + 0) * SH + k] = f2bf(v.x);
    wt[(n + 1) * SH + k] = f2bf(v.y);
    wt[(n + 2) * SH + k] = f2bf(v.z);
    wt[(n + 3) * SH + k] = f2bf(v.w);
  }
}

__global__ __launch_bounds__(NTHREADS, 1)
void meganerf_moe(const float* __restrict__ x,  const float* __restrict__ cen,
                  const float* __restrict__ W1, const float* __restrict__ b1,
                  const float* __restrict__ W2, const float* __restrict__ b2,
                  const float* __restrict__ W3, const float* __restrict__ b3,
                  const float* __restrict__ Wo, const float* __restrict__ bo,
                  float* __restrict__ out) {
  // ~302 KB of the 320 KB WGP LDS
  __shared__ __align__(16) unsigned short sFeat[TILE * SF];  //  26.6 KB
  __shared__ __align__(16) unsigned short sHA[TILE * SH];    //  67.6 KB
  __shared__ __align__(16) unsigned short sHB[TILE * SH];    //  67.6 KB
  __shared__ __align__(16) unsigned short sW[HD * SH];       // 135.2 KB
  __shared__ float sBias[HD];
  __shared__ float sWo[HD * OUT_D];
  __shared__ float sBo[OUT_D];
  __shared__ float sGate[TILE];
  __shared__ int   sAny;

  const int tid = threadIdx.x;
  const int e   = blockIdx.y;
  const int n0  = blockIdx.x * TILE;

  if (tid == 0) sAny = 0;
  __syncthreads();

  // ---- gating weights for this expert over the tile ----
  if (tid < TILE) {
    const int n = n0 + tid;
    const float px = x[(size_t)n * XROW + 1];
    const float py = x[(size_t)n * XROW + 2];
    float d[NUM_E], mind = 3.4e38f;
#pragma unroll
    for (int j = 0; j < NUM_E; ++j) {
      const float dx = px - cen[j * 3 + 1];
      const float dy = py - cen[j * 3 + 2];
      d[j] = sqrtf(dx * dx + dy * dy);
      mind = fminf(mind, d[j]);
    }
    float s = 0.f, we = 0.f;
#pragma unroll
    for (int j = 0; j < NUM_E; ++j) {
      const float inv = (d[j] > MARGIN_F * mind) ? 0.f : 1.f / (d[j] + 1e-8f);
      s += inv;
      if (j == e) we = inv;
    }
    const float g = we / s;
    sGate[tid] = g;
    if (g != 0.f) atomicOr(&sAny, 1);
  }
  __syncthreads();
  if (sAny == 0) return;   // whole tile gated off for this expert

  // ---- stage feats tile (bf16, K zero-padded 90 -> 96) + head weights ----
  for (int i = tid; i < TILE * K1P; i += NTHREADS) {
    const int row = i / K1P, col = i - row * K1P;
    const float v = (col < IN_F) ? x[(size_t)(n0 + row) * XROW + 3 + col] : 0.f;
    sFeat[row * SF + col] = f2bf(v);
  }
  for (int i = tid; i < HD * OUT_D; i += NTHREADS)
    sWo[i] = Wo[(size_t)e * HD * OUT_D + i];
  if (tid < OUT_D) sBo[tid] = bo[e * OUT_D + tid];

  // ---- layer 1 ----
  load_wT(W1 + (size_t)e * IN_F * HD, IN_F, K1P, sW, tid);
  for (int i = tid; i < HD; i += NTHREADS) sBias[i] = b1[(size_t)e * HD + i];
  __syncthreads();
  mlp_layer(sFeat, SF, K1P, sW, sBias, sHA, tid, true);
  __syncthreads();

  // ---- layer 2 ----
  load_wT(W2 + (size_t)e * HD * HD, HD, HD, sW, tid);
  for (int i = tid; i < HD; i += NTHREADS) sBias[i] = b2[(size_t)e * HD + i];
  __syncthreads();
  mlp_layer(sHA, SH, HD, sW, sBias, sHB, tid, true);
  __syncthreads();

  // ---- layer 3 ----
  load_wT(W3 + (size_t)e * HD * HD, HD, HD, sW, tid);
  for (int i = tid; i < HD; i += NTHREADS) sBias[i] = b3[(size_t)e * HD + i];
  __syncthreads();
  mlp_layer(sHB, SH, HD, sW, sBias, sHA, tid, true);
  __syncthreads();

  // ---- output head (OUT=4, VALU) + gated accumulation ----
  const int row = tid >> 2;       // 512 threads = 128 rows x 4 outputs
  const int o   = tid & 3;
  float acc = 0.f;
  for (int h = 0; h < HD; ++h)
    acc = fmaf(bf2f(sHA[row * SH + h]), sWo[h * OUT_D + o], acc);
  const float g = sGate[row];
  if (g != 0.f)
    atomicAdd(out + (size_t)(n0 + row) * OUT_D + o, g * (acc + sBo[o]));
}

__global__ void zero_out_kernel(float* __restrict__ p, int n) {
  const int i = blockIdx.x * blockDim.x + threadIdx.x;
  if (i < n) p[i] = 0.f;
}

extern "C" void kernel_launch(void* const* d_in, const int* in_sizes, int n_in,
                              void* d_out, int out_size, void* d_ws, size_t ws_size,
                              hipStream_t stream) {
  (void)in_sizes; (void)n_in; (void)d_ws; (void)ws_size; (void)out_size;
  const float* x   = (const float*)d_in[0];
  const float* cen = (const float*)d_in[1];
  const float* W1  = (const float*)d_in[2];
  const float* b1  = (const float*)d_in[3];
  const float* W2  = (const float*)d_in[4];
  const float* b2  = (const float*)d_in[5];
  const float* W3  = (const float*)d_in[6];
  const float* b3  = (const float*)d_in[7];
  const float* Wo  = (const float*)d_in[8];
  const float* bo  = (const float*)d_in[9];
  float* out = (float*)d_out;

  const int outN = NPTS * OUT_D;
  zero_out_kernel<<<(outN + NTHREADS - 1) / NTHREADS, NTHREADS, 0, stream>>>(out, outN);

  dim3 grid(NPTS / TILE, NUM_E);
  meganerf_moe<<<grid, NTHREADS, 0, stream>>>(x, cen, W1, b1, W2, b2, W3, b3, Wo, bo, out);
}
